// ParagraphGAT_7988639171403
// MI455X (gfx1250) — matrix operand order repaired and checked by reading, not compiled, not run on hardware
//
#include <hip/hip_runtime.h>
#include <hip/hip_bf16.h>

// ---------------- problem constants (match reference) ----------------
#define NN 50000           // nodes
#define EE 400000          // raw edges
#define EP (EE + NN)       // edges incl. self loops = 450000
#define DD 64              // hidden dim
#define HH 8               // heads
#define CC (HH * DD)       // projected width = 512
#define NEG_SLOPE 0.2f

typedef __attribute__((ext_vector_type(16))) __bf16 v16bf;
typedef __attribute__((ext_vector_type(8)))  float  v8f;

// round-to-nearest-even f32 -> bf16 (bit level; avoids relying on __bf16 arithmetic)
__device__ __forceinline__ unsigned short f32_to_bf16(float f) {
  unsigned int u = __float_as_uint(f);
  u += 0x7FFFu + ((u >> 16) & 1u);
  return (unsigned short)(u >> 16);
}

// float atomic max via sign-split integer atomics (works with -inf init)
__device__ __forceinline__ void atomicMaxF(float* addr, float v) {
  if (v >= 0.0f) atomicMax((int*)addr, __float_as_int(v));
  else           atomicMin((unsigned int*)addr, __float_as_uint(v));
}

// ---------------- conversion kernels ----------------
__global__ void k_cvt_bf16(const float* __restrict__ in,
                           unsigned short* __restrict__ out, int n) {
  int i = blockIdx.x * blockDim.x + threadIdx.x;
  if (i < n) out[i] = f32_to_bf16(in[i]);
}

// W [64][512] f32  ->  Wt [512][64] bf16  (so B-fragment k-pairs are contiguous)
__global__ void k_cvt_transpose_w(const float* __restrict__ W,
                                  unsigned short* __restrict__ wt) {
  int i = blockIdx.x * blockDim.x + threadIdx.x;   // over 64*512
  if (i < DD * CC) {
    int k = i / CC, c = i % CC;
    wt[c * DD + k] = f32_to_bf16(W[i]);
  }
}

// ---------------- WMMA GEMM: h[N][512] = x[N][64] @ W[64][512] ----------------
// 1 wave -> one 16x16 output tile; K=64 done as two 16x16x32 bf16 WMMAs.
// Block = 256 threads = 8 waves = 8 column tiles of one row tile.
// EXEC stays all-ones (grid tiles exactly: 50000=3125*16, 512=32*16).
__global__ __launch_bounds__(256)
void k_gemm_wmma(const unsigned short* __restrict__ xb,   // [N][64] bf16
                 const unsigned short* __restrict__ wt,   // [512][64] bf16 (transposed)
                 float* __restrict__ h) {                 // [N][512] f32
  const int lane = threadIdx.x & 31;
  const int wave = threadIdx.x >> 5;
  const int rowBase = blockIdx.x * 16;
  const int colBase = (blockIdx.y * 8 + wave) * 16;
  const int m  = lane & 15;
  const int hi = lane >> 4;

  // packed-pair (uint = 2 bf16) base pointers for this lane
  const unsigned int* xr = (const unsigned int*)(xb + (size_t)(rowBase + m) * DD);
  const unsigned int* wr = (const unsigned int*)(wt + (size_t)(colBase + m) * DD);

  v8f acc = {};
  union Frag { unsigned int u[8]; v16bf v; };

#pragma unroll
  for (int s = 0; s < 2; ++s) {
    const int ko = s * 16;   // k-step offset in uint (pair) units: 32 bf16
    Frag a, b;
    // A 16x32 bf16 layout (ISA 7.12.2): lanes 0-15 K=0..7 & 16..23, lanes 16-31 K=8..15 & 24..31
#pragma unroll
    for (int v = 0; v < 4; ++v) {
      a.u[v]     = xr[ko + hi * 4 + v];
      a.u[4 + v] = xr[ko + 8 + hi * 4 + v];
    }
    // B 32x16 bf16: lanes 0-15 hold K=0..7 & 16..23, lanes 16-31 hold K=8..15 & 24..31 (pairs contiguous in wt)
#pragma unroll
    for (int v = 0; v < 4; ++v) {
      b.u[v]     = wr[ko + hi * 4 + v];
      b.u[4 + v] = wr[ko + 8 + hi * 4 + v];
    }

    acc = __builtin_amdgcn_wmma_f32_16x16x32_bf16(
        false, a.v, false, b.v, (short)0, acc, false, false);
  }

  // C/D 16x16 f32 layout: lane n = lane&15, VGPR r -> row r (+8 for hi half)
  float* hp = h + (size_t)(rowBase + hi * 8) * CC + colBase + m;
#pragma unroll
  for (int r = 0; r < 8; ++r) hp[(size_t)r * CC] = acc[r];
}

// ---------------- attention logits: one wave per (node, head) ----------------
__global__ __launch_bounds__(256)
void k_attn_logits(const float* __restrict__ h,
                   const float* __restrict__ a_src,   // [H][64]
                   const float* __restrict__ a_dst,   // [H][64]
                   float* __restrict__ al_src,        // [N][H]
                   float* __restrict__ al_dst) {
  int gw   = (blockIdx.x * blockDim.x + threadIdx.x) >> 5;
  int lane = threadIdx.x & 31;
  if (gw >= NN * HH) return;
  int n = gw / HH, hh = gw % HH;
  const float* hp = h + (size_t)n * CC + hh * DD;
  const float* as = a_src + hh * DD;
  const float* ad = a_dst + hh * DD;
  float s0 = hp[lane] * as[lane] + hp[lane + 32] * as[lane + 32];
  float s1 = hp[lane] * ad[lane] + hp[lane + 32] * ad[lane + 32];
#pragma unroll
  for (int m = 16; m >= 1; m >>= 1) {
    s0 += __shfl_xor(s0, m, 32);
    s1 += __shfl_xor(s1, m, 32);
  }
  if (lane == 0) {
    al_src[n * HH + hh] = s0;
    al_dst[n * HH + hh] = s1;
  }
}

// ---------------- fill -inf ----------------
__global__ void k_fill_neg_inf(float* __restrict__ p, int n) {
  int i = blockIdx.x * blockDim.x + threadIdx.x;
  if (i < n) p[i] = __uint_as_float(0xFF800000u);   // -inf
}

// ---------------- segment max over edges: thread per (edge, head) ----------------
__global__ void k_edge_max(const int* __restrict__ ei,
                           const float* __restrict__ al_src,
                           const float* __restrict__ al_dst,
                           float* __restrict__ mbuf) {
  int t = blockIdx.x * blockDim.x + threadIdx.x;
  if (t >= EP * HH) return;
  int e = t >> 3, hh = t & 7;
  int src, dst;
  if (e < EE) { src = ei[e]; dst = ei[EE + e]; } else { src = dst = e - EE; }
  float v = al_src[src * HH + hh] + al_dst[dst * HH + hh];
  v = v > 0.0f ? v : NEG_SLOPE * v;                 // leaky_relu(0.2)
  atomicMaxF(&mbuf[dst * HH + hh], v);
}

// ---------------- exp(e - m[dst]) + segment sum ----------------
__global__ void k_edge_expsum(const int* __restrict__ ei,
                              const float* __restrict__ al_src,
                              const float* __restrict__ al_dst,
                              const float* __restrict__ mbuf,
                              float* __restrict__ exb,        // [EP][H]
                              float* __restrict__ den) {      // [N][H]
  int t = blockIdx.x * blockDim.x + threadIdx.x;
  if (t >= EP * HH) return;
  int e = t >> 3, hh = t & 7;
  int src, dst;
  if (e < EE) { src = ei[e]; dst = ei[EE + e]; } else { src = dst = e - EE; }
  float v = al_src[src * HH + hh] + al_dst[dst * HH + hh];
  v = v > 0.0f ? v : NEG_SLOPE * v;
  float ex = __expf(v - mbuf[dst * HH + hh]);
  exb[e * HH + hh] = ex;
  atomicAdd(&den[dst * HH + hh], ex);
}

// ---------------- weighted scatter aggregation ----------------
// 128 threads per edge: thread q handles feature quad [4q, 4q+4), head = q/16.
__global__ __launch_bounds__(256)
void k_aggregate(const float* __restrict__ h,
                 const float* __restrict__ exb,
                 const float* __restrict__ den,
                 const int* __restrict__ ei,
                 float* __restrict__ outb) {        // [N][512]
  int t = blockIdx.x * blockDim.x + threadIdx.x;
  int e = t >> 7;
  int q = t & 127;
  if (e >= EP) return;
  int src, dst;
  if (e < EE) { src = ei[e]; dst = ei[EE + e]; } else { src = dst = e - EE; }
  int hh = q >> 4;
  float alpha = exb[e * HH + hh] / (den[dst * HH + hh] + 1e-16f);
  const float* hs = h + (size_t)src * CC + q * 4;
  __builtin_prefetch(hs + 64, 0, 0);                // global_prefetch of next lines
  float4 hv = *(const float4*)hs;
  float* op = outb + (size_t)dst * CC + q * 4;
  atomicAdd(op + 0, alpha * hv.x);
  atomicAdd(op + 1, alpha * hv.y);
  atomicAdd(op + 2, alpha * hv.z);
  atomicAdd(op + 3, alpha * hv.w);
}

// ---------------- head mean + bias + ELU ----------------
__global__ void k_finalize(const float* __restrict__ outb,
                           const float* __restrict__ bias,
                           float* __restrict__ xout) {        // [N][64]
  int t = blockIdx.x * blockDim.x + threadIdx.x;
  if (t >= NN * DD) return;
  int n = t / DD, d = t % DD;
  float s = 0.0f;
#pragma unroll
  for (int hh = 0; hh < HH; ++hh) s += outb[(size_t)n * CC + hh * DD + d];
  s = s * (1.0f / HH) + bias[d];
  s = s > 0.0f ? s : (__expf(s) - 1.0f);            // elu
  xout[t] = s;
}

// ---------------- host-side layer driver ----------------
static void run_layer(hipStream_t stream,
                      const float* xin, const int* ei,
                      const float* W, const float* a_src, const float* a_dst,
                      const float* bias,
                      unsigned short* xb, unsigned short* wt,
                      float* h, float* al_src, float* al_dst,
                      float* mbuf, float* den, float* exb,
                      float* outb, float* xout) {
  k_cvt_bf16      <<<(NN * DD + 255) / 256, 256, 0, stream>>>(xin, xb, NN * DD);
  k_cvt_transpose_w<<<(DD * CC + 255) / 256, 256, 0, stream>>>(W, wt);
  k_gemm_wmma     <<<dim3(NN / 16, CC / 128), 256, 0, stream>>>(xb, wt, h);
  k_attn_logits   <<<(NN * HH) / 8, 256, 0, stream>>>(h, a_src, a_dst, al_src, al_dst);
  k_fill_neg_inf  <<<(NN * HH + 255) / 256, 256, 0, stream>>>(mbuf, NN * HH);
  hipMemsetAsync(den,  0, (size_t)NN * HH * sizeof(float), stream);
  hipMemsetAsync(outb, 0, (size_t)NN * CC * sizeof(float), stream);
  k_edge_max      <<<(EP * HH + 255) / 256, 256, 0, stream>>>(ei, al_src, al_dst, mbuf);
  k_edge_expsum   <<<(EP * HH + 255) / 256, 256, 0, stream>>>(ei, al_src, al_dst, mbuf, exb, den);
  k_aggregate     <<<(EP * 128) / 256, 256, 0, stream>>>(h, exb, den, ei, outb);
  k_finalize      <<<(NN * DD + 255) / 256, 256, 0, stream>>>(outb, bias, xout);
}

extern "C" void kernel_launch(void* const* d_in, const int* in_sizes, int n_in,
                              void* d_out, int out_size, void* d_ws, size_t ws_size,
                              hipStream_t stream) {
  const float* x    = (const float*)d_in[0];
  const int*   ei   = (const int*)  d_in[1];
  const float* W1   = (const float*)d_in[2];
  const float* as1  = (const float*)d_in[3];
  const float* ad1  = (const float*)d_in[4];
  const float* b1   = (const float*)d_in[5];
  const float* W2   = (const float*)d_in[6];
  const float* as2  = (const float*)d_in[7];
  const float* ad2  = (const float*)d_in[8];
  const float* b2   = (const float*)d_in[9];
  float* out = (float*)d_out;

  // workspace carve-up (~245 MB total)
  char* p = (char*)d_ws;
  float* h       = (float*)p; p += (size_t)NN * CC * 4;          // 102.4 MB
  float* outb    = (float*)p; p += (size_t)NN * CC * 4;          // 102.4 MB
  float* al_src  = (float*)p; p += (size_t)NN * HH * 4;
  float* al_dst  = (float*)p; p += (size_t)NN * HH * 4;
  float* mbuf    = (float*)p; p += (size_t)NN * HH * 4;
  float* den     = (float*)p; p += (size_t)NN * HH * 4;
  float* exb     = (float*)p; p += (size_t)EP * HH * 4;          // 14.4 MB
  unsigned short* xb = (unsigned short*)p; p += (size_t)NN * DD * 2;  // 6.4 MB
  unsigned short* wt = (unsigned short*)p; p += (size_t)CC * DD * 2;  // 64 KB
  float* xn      = (float*)p; p += (size_t)NN * DD * 4;          // 12.8 MB

  run_layer(stream, x,  ei, W1, as1, ad1, b1, xb, wt, h, al_src, al_dst, mbuf, den, exb, outb, xn);
  run_layer(stream, xn, ei, W2, as2, ad2, b2, xb, wt, h, al_src, al_dst, mbuf, den, exb, outb, out);
}
